// ShiftedWindowAttn_90486370992998
// MI455X (gfx1250) — matrix operand order repaired and checked by reading, not compiled, not run on hardware
//
#include <hip/hip_runtime.h>

#define WSZ 7
#define SHIFT_ 3
#define NH_ 6
#define HD_ 32
#define C_ 192
#define C3_ 576
#define HH_ 56
#define NW_ 64
#define L_ 49
#define LP_ 64

typedef __bf16 bf16_t;
typedef bf16_t bf16x16 __attribute__((ext_vector_type(16)));
typedef float v8f __attribute__((ext_vector_type(8)));

union FragU { bf16x16 v; uint4 q[2]; };

// Native bf16 convert (RNE) -> raw bits; lowers to hardware cvt on gfx1250.
__device__ __forceinline__ unsigned short f2bfbits(float f) {
  bf16_t h = (bf16_t)f;
  return __builtin_bit_cast(unsigned short, h);
}

// A-fragment (M x K = 16x32 bf16) from row-major storage (pitch in elements).
// lanes 0-15: row=lane, K chunks {0..7, 16..23}; lanes 16-31: K chunks {8..15, 24..31}.
__device__ __forceinline__ bf16x16 ldA(const bf16_t* base, int row, int kB, int pitch, int lane) {
  int r = row + (lane & 15);
  int k = kB + ((lane >> 4) << 3);
  const bf16_t* p = base + r * pitch + k;
  FragU f;
  f.q[0] = *reinterpret_cast<const uint4*>(p);
  f.q[1] = *reinterpret_cast<const uint4*>(p + 16);
  return f.v;
}

// B-fragment (K x N = 32x16 bf16) from transposed storage Bt[n][k] contiguous in k.
// lane n (<16): column n, K chunks {0..7, 16..23}; lane n+16: K chunks {8..15, 24..31}.
__device__ __forceinline__ bf16x16 ldB(const bf16_t* base, int nrow, int kB, int pitch, int lane) {
  int r = nrow + (lane & 15);
  int k = kB + ((lane >> 4) << 3);
  const bf16_t* p = base + r * pitch + k;
  FragU f;
  f.q[0] = *reinterpret_cast<const uint4*>(p);
  f.q[1] = *reinterpret_cast<const uint4*>(p + 16);
  return f.v;
}

#define WMMA_BF16(a, b, c) \
  __builtin_amdgcn_wmma_f32_16x16x32_bf16(false, (a), false, (b), (short)0, (c), false, false)

// ---------------- prep: weight transpose->bf16, padded bias & mask tables ----------------
__global__ void swin_prep(const float* __restrict__ qkv_w,
                          const float* __restrict__ proj_w,
                          const float* __restrict__ rel_table,
                          unsigned short* __restrict__ wqkvT,
                          unsigned short* __restrict__ wpT,
                          float* __restrict__ biasPad,
                          float* __restrict__ maskPad) {
  int idx = blockIdx.x * 256 + threadIdx.x;
  const int N1 = C3_ * C_;          // 110592
  const int N2 = C_ * C_;           // 36864
  const int N3 = NH_ * LP_ * LP_;   // 24576
  const int N4 = NW_ * LP_ * LP_;   // 262144
  if (idx < N1) {
    int n = idx / C_, k = idx - n * C_;
    wqkvT[idx] = f2bfbits(qkv_w[k * C3_ + n]);
    return;
  }
  idx -= N1;
  if (idx < N2) {
    int n = idx / C_, k = idx - n * C_;
    wpT[idx] = f2bfbits(proj_w[k * C_ + n]);
    return;
  }
  idx -= N2;
  if (idx < N3) {
    int h = idx >> 12, rem = idx & 4095, i = rem >> 6, j = rem & 63;
    float v;
    if (j >= L_) v = -3.0e38f;          // kill padded key columns in softmax
    else if (i >= L_) v = 0.0f;         // padded query rows: harmless
    else {
      int rid = ((i / 7) - (j / 7) + 6) * 13 + ((i % 7) - (j % 7) + 6);
      v = rel_table[rid * NH_ + h];
    }
    biasPad[idx] = v;
    return;
  }
  idx -= N3;
  if (idx < N4) {
    int w = idx >> 12, rem = idx & 4095, i = rem >> 6, j = rem & 63;
    float v = 0.0f;
    if (i < L_ && j < L_) {
      int wh = w >> 3, ww = w & 7;
      int hi = wh * 7 + i / 7, wi = ww * 7 + i % 7;
      int hj = wh * 7 + j / 7, wj = ww * 7 + j % 7;
      int ri = (hi < 49 ? 0 : (hi < 53 ? 1 : 2)) * 3 + (wi < 49 ? 0 : (wi < 53 ? 1 : 2));
      int rj = (hj < 49 ? 0 : (hj < 53 ? 1 : 2)) * 3 + (wj < 49 ? 0 : (wj < 53 ? 1 : 2));
      v = (ri != rj) ? -100.0f : 0.0f;
    }
    maskPad[idx] = v;
  }
}

// ---------------- fused shifted-window attention: 1 block = 1 (batch, window) ----------------
__global__ __launch_bounds__(256) void swin_attn(
    const float* __restrict__ xg,
    const float* __restrict__ qkv_b,
    const float* __restrict__ proj_b,
    const unsigned short* __restrict__ wqkvT_,
    const unsigned short* __restrict__ wpT_,
    const float* __restrict__ biasPad,
    const float* __restrict__ maskPad,
    float* __restrict__ outg) {
  extern __shared__ char smem[];
  bf16_t* Xb = (bf16_t*)smem;                              // 64x192 bf16 (reused as O)
  unsigned short* XbS = (unsigned short*)smem;
  bf16_t* Qb = (bf16_t*)(smem + 24576);
  unsigned short* QbS = (unsigned short*)(smem + 24576);
  bf16_t* Kb = (bf16_t*)(smem + 49152);
  unsigned short* KbS = (unsigned short*)(smem + 49152);
  bf16_t* Vt = (bf16_t*)(smem + 73728);                    // V transposed: 192 x 64
  unsigned short* VtS = (unsigned short*)(smem + 73728);
  unsigned short* Ps = (unsigned short*)(smem + 98304);    // 8 waves x (16x64) P scratch

  const bf16_t* wqkvT = (const bf16_t*)wqkvT_;
  const bf16_t* wpT = (const bf16_t*)wpT_;

  int tid = threadIdx.x, wave = tid >> 5, lane = tid & 31;
  int blk = blockIdx.x;
  int wdw = blk & 63;
  int bb = blk >> 6;
  int wh = wdw >> 3, ww = wdw & 7;
  int hi8 = (lane >> 4) << 3;
  int col = lane & 15;

  // Warm the cache for the first weight tiles this wave will consume in GEMM1
  // (lowers to global_prefetch_b8); overlaps with the stage-0 LDS fill below.
  {
    const char* pfb = (const char*)(wqkvT + (size_t)((wave >> 2) * 18 * 16) * C_);
    __builtin_prefetch(pfb + (size_t)lane * 128, 0, 1);
    __builtin_prefetch(pfb + 4096 + (size_t)lane * 128, 0, 1);
  }

  // ---- stage 0: rolled window -> LDS bf16 (rows 49..63 zeroed) ----
  for (int idx = tid; idx < 360; idx += 256) {
    int row = 49 + idx / 24, c8 = (idx % 24) << 3;
    *reinterpret_cast<uint4*>(XbS + row * C_ + c8) = make_uint4(0u, 0u, 0u, 0u);
  }
  for (int idx = tid; idx < 2352; idx += 256) {
    int i = idx / 48, c4 = (idx % 48) << 2;
    int ti = i / 7, tj = i % 7;
    int hs = wh * 7 + ti + SHIFT_; if (hs >= HH_) hs -= HH_;
    int ws = ww * 7 + tj + SHIFT_; if (ws >= HH_) ws -= HH_;
    const float4 xv = *reinterpret_cast<const float4*>(
        xg + ((((size_t)bb * HH_ + hs) * HH_ + ws) * C_ + c4));
    ushort4 hv;
    hv.x = f2bfbits(xv.x); hv.y = f2bfbits(xv.y);
    hv.z = f2bfbits(xv.z); hv.w = f2bfbits(xv.w);
    *reinterpret_cast<ushort4*>(XbS + i * C_ + c4) = hv;
  }
  __syncthreads();

  // ---- stage 1: QKV GEMM (64x192 @ 192x576), epilogue scatters Q,K row-major and V transposed ----
  {
    int mt = wave & 3, half = wave >> 2;
    bf16x16 aq[6];
#pragma unroll
    for (int ks = 0; ks < 6; ++ks) aq[ks] = ldA(Xb, mt * 16, ks * 32, C_, lane);
    for (int t = 0; t < 18; ++t) {
      int nt = half * 18 + t;
      v8f acc = {};
#pragma unroll
      for (int ks = 0; ks < 6; ++ks) {
        bf16x16 b = ldB(wqkvT, nt * 16, ks * 32, C_, lane);
        acc = WMMA_BF16(aq[ks], b, acc);
      }
      int nG = nt * 16 + col;
      float bias = qkv_b[nG];
      if (nt < 12) {        // Q, pre-scaled by HD^-0.5
#pragma unroll
        for (int r = 0; r < 8; ++r) {
          int row = mt * 16 + r + hi8;
          QbS[row * C_ + nG] = f2bfbits((acc[r] + bias) * 0.17677669529663689f);
        }
      } else if (nt < 24) { // K
        int c = nG - C_;
#pragma unroll
        for (int r = 0; r < 8; ++r) {
          int row = mt * 16 + r + hi8;
          KbS[row * C_ + c] = f2bfbits(acc[r] + bias);
        }
      } else {              // V transposed
        int ch = nG - 2 * C_;
#pragma unroll
        for (int r = 0; r < 8; ++r) {
          int row = mt * 16 + r + hi8;
          VtS[ch * LP_ + row] = f2bfbits(acc[r] + bias);
        }
      }
    }
  }
  __syncthreads();

  // ---- stage 2: per (head, M-tile) row-block: S=QK^T, bias+mask, softmax in regs, P@V ----
  unsigned short* ObS = XbS;  // X is dead; reuse as attention output O (64x192 bf16)
  for (int rb = wave; rb < 24; rb += 8) {
    int h = rb >> 2, mt = rb & 3, mBase = mt * 16;
    bf16x16 a = ldA(Qb, mBase, h * HD_, C_, lane);
    v8f s[4];
#pragma unroll
    for (int n = 0; n < 4; ++n) {
      bf16x16 b = ldB(Kb, n * 16, h * HD_, C_, lane);
      v8f z = {};
      s[n] = WMMA_BF16(a, b, z);
    }
    const float* bp = biasPad + h * 4096;
    const float* mp = maskPad + (size_t)wdw * 4096;
#pragma unroll
    for (int n = 0; n < 4; ++n)
#pragma unroll
      for (int r = 0; r < 8; ++r) {
        int i = mBase + r + hi8, j = n * 16 + col;
        s[n][r] += bp[i * 64 + j] + mp[i * 64 + j];
      }
    // softmax over j: rows live in a fixed vgpr across one 16-lane half + 4 fragments
#pragma unroll
    for (int r = 0; r < 8; ++r) {
      float m = fmaxf(fmaxf(s[0][r], s[1][r]), fmaxf(s[2][r], s[3][r]));
      m = fmaxf(m, __shfl_xor(m, 1, 32));
      m = fmaxf(m, __shfl_xor(m, 2, 32));
      m = fmaxf(m, __shfl_xor(m, 4, 32));
      m = fmaxf(m, __shfl_xor(m, 8, 32));
      float sum = 0.f;
#pragma unroll
      for (int n = 0; n < 4; ++n) { float e = __expf(s[n][r] - m); s[n][r] = e; sum += e; }
      sum += __shfl_xor(sum, 1, 32);
      sum += __shfl_xor(sum, 2, 32);
      sum += __shfl_xor(sum, 4, 32);
      sum += __shfl_xor(sum, 8, 32);
      float inv = 1.0f / sum;
#pragma unroll
      for (int n = 0; n < 4; ++n) s[n][r] *= inv;
    }
    // D-layout -> A-layout via per-wave LDS scratch
    unsigned short* Pw = Ps + wave * 1024;
#pragma unroll
    for (int n = 0; n < 4; ++n)
#pragma unroll
      for (int r = 0; r < 8; ++r)
        Pw[(r + hi8) * 64 + n * 16 + col] = f2bfbits(s[n][r]);
    bf16x16 p0 = ldA((const bf16_t*)Pw, 0, 0, 64, lane);
    bf16x16 p1 = ldA((const bf16_t*)Pw, 0, 32, 64, lane);
#pragma unroll
    for (int n2 = 0; n2 < 2; ++n2) {
      v8f o = {};
      o = WMMA_BF16(p0, ldB(Vt, h * HD_ + n2 * 16, 0, LP_, lane), o);
      o = WMMA_BF16(p1, ldB(Vt, h * HD_ + n2 * 16, 32, LP_, lane), o);
#pragma unroll
      for (int r = 0; r < 8; ++r) {
        int row = mBase + r + hi8;
        ObS[row * C_ + h * HD_ + n2 * 16 + col] = f2bfbits(o[r]);
      }
    }
  }
  __syncthreads();

  // ---- stage 3: projection GEMM (64x192 @ 192x192) + un-shift scatter to global ----
  const bf16_t* Ob = (const bf16_t*)ObS;
  for (int tt = 0; tt < 6; ++tt) {
    int t = wave * 6 + tt, mt = t / 12, nt = t % 12;
    v8f acc = {};
#pragma unroll
    for (int ks = 0; ks < 6; ++ks) {
      acc = WMMA_BF16(ldA(Ob, mt * 16, ks * 32, C_, lane),
                      ldB(wpT, nt * 16, ks * 32, C_, lane), acc);
    }
    int c = nt * 16 + col;
    float pb = proj_b[c];
#pragma unroll
    for (int r = 0; r < 8; ++r) {
      int m = mt * 16 + r + hi8;
      if (m < L_) {
        int ti = m / 7, tj = m % 7;
        int ho = wh * 7 + ti + SHIFT_; if (ho >= HH_) ho -= HH_;
        int wo = ww * 7 + tj + SHIFT_; if (wo >= HH_) wo -= HH_;
        outg[(((size_t)bb * HH_ + ho) * HH_ + wo) * C_ + c] = acc[r] + pb;
      }
    }
  }
}

extern "C" void kernel_launch(void* const* d_in, const int* in_sizes, int n_in,
                              void* d_out, int out_size, void* d_ws, size_t ws_size,
                              hipStream_t stream) {
  const float* x = (const float*)d_in[0];
  const float* qkv_w = (const float*)d_in[1];
  const float* qkv_b = (const float*)d_in[2];
  const float* proj_w = (const float*)d_in[3];
  const float* proj_b = (const float*)d_in[4];
  const float* rel_table = (const float*)d_in[5];

  char* ws = (char*)d_ws;
  unsigned short* wqkvT = (unsigned short*)ws;            // 576*192*2   = 221184 B
  unsigned short* wpT = (unsigned short*)(ws + 221184);   // 192*192*2   =  73728 B
  float* biasPad = (float*)(ws + 294912);                 // 6*64*64*4   =  98304 B
  float* maskPad = (float*)(ws + 393216);                 // 64*64*64*4  = 1048576 B

  const int NPREP = 110592 + 36864 + 24576 + 262144;
  hipLaunchKernelGGL(swin_prep, dim3((NPREP + 255) / 256), dim3(256), 0, stream,
                     qkv_w, proj_w, rel_table, wqkvT, wpT, biasPad, maskPad);

  const int SMEM = 114688;  // 4*24576 + 16384 bytes of LDS (<< 320KB WGP limit)
  hipFuncSetAttribute(reinterpret_cast<const void*>(&swin_attn),
                      hipFuncAttributeMaxDynamicSharedMemorySize, SMEM);
  hipLaunchKernelGGL(swin_attn, dim3(64 * NW_), dim3(256), SMEM, stream,
                     x, qkv_b, proj_b, wqkvT, wpT, biasPad, maskPad, (float*)d_out);
}
